// NON_ZERO_RMSELoss_Spatial_AE_51891794870959
// MI455X (gfx1250) — matrix operand order repaired and checked by reading, not compiled, not run on hardware
//
#include <hip/hip_runtime.h>

// Problem dimensions (match the reference)
#define BDIM 8192
#define SDIM 4096
#define EPSV 1e-6f

typedef __bf16 bf16_t;
typedef bf16_t v16bf __attribute__((ext_vector_type(16)));
typedef float  v8f   __attribute__((ext_vector_type(8)));

// ---------------------------------------------------------------------------
// helpers
// ---------------------------------------------------------------------------
__device__ __forceinline__ unsigned short f2bf(float f) {
    // round-to-nearest-even f32 -> bf16 (bits)
    unsigned int u = __float_as_uint(f);
    unsigned int r = u + 0x7FFFu + ((u >> 16) & 1u);
    return (unsigned short)(r >> 16);
}

// Load a 16-element bf16 fragment as two contiguous 16-byte chunks.
__device__ __forceinline__ v16bf load_frag(const unsigned short* lo,
                                           const unsigned short* hi) {
    union { uint4 q[2]; v16bf v; } u;
    u.q[0] = *reinterpret_cast<const uint4*>(lo);
    u.q[1] = *reinterpret_cast<const uint4*>(hi);
    return u.v;
}

// ---------------------------------------------------------------------------
// 1) sq = (y - yhat)^2 -> bf16; accumulate sum(mask*sq) and sum(mask)
//    one float4 per thread (B*S/4 threads total)
// ---------------------------------------------------------------------------
__global__ __launch_bounds__(256)
void prep_sq_mask(const float* __restrict__ yhat, const float* __restrict__ y,
                  unsigned short* __restrict__ sqB, float* __restrict__ acc) {
    size_t i = (size_t)blockIdx.x * blockDim.x + threadIdx.x;
    const float4 y4 = reinterpret_cast<const float4*>(y)[i];
    const float4 p4 = reinterpret_cast<const float4*>(yhat)[i];

    float d0 = y4.x - p4.x, d1 = y4.y - p4.y, d2 = y4.z - p4.z, d3 = y4.w - p4.w;
    float s0 = d0 * d0, s1 = d1 * d1, s2 = d2 * d2, s3 = d3 * d3;

    ushort4 o;
    o.x = f2bf(s0); o.y = f2bf(s1); o.z = f2bf(s2); o.w = f2bf(s3);
    reinterpret_cast<ushort4*>(sqB)[i] = o;

    float m0 = (y4.x != 0.f) ? 1.f : 0.f;
    float m1 = (y4.y != 0.f) ? 1.f : 0.f;
    float m2 = (y4.z != 0.f) ? 1.f : 0.f;
    float m3 = (y4.w != 0.f) ? 1.f : 0.f;
    float msum = m0 + m1 + m2 + m3;
    float qsum = m0 * s0 + m1 * s1 + m2 * s2 + m3 * s3;

    // wave32 reduce, then LDS across the 8 waves, one atomic per block
    for (int off = 16; off > 0; off >>= 1) {
        msum += __shfl_down(msum, off, 32);
        qsum += __shfl_down(qsum, off, 32);
    }
    __shared__ float sm[8], sq[8];
    int lane = threadIdx.x & 31, wave = threadIdx.x >> 5;
    if (lane == 0) { sm[wave] = msum; sq[wave] = qsum; }
    __syncthreads();
    if (threadIdx.x == 0) {
        float tm = 0.f, tq = 0.f;
        for (int w = 0; w < 8; ++w) { tm += sm[w]; tq += sq[w]; }
        atomicAdd(acc + 0, tq);
        atomicAdd(acc + 1, tm);
    }
}

// ---------------------------------------------------------------------------
// 2) W f32 -> bf16 (row-major, W[i][j] with j contiguous)
// ---------------------------------------------------------------------------
__global__ __launch_bounds__(256)
void conv_w(const float* __restrict__ W, unsigned short* __restrict__ Wb) {
    size_t i = (size_t)blockIdx.x * blockDim.x + threadIdx.x;
    const float4 w4 = reinterpret_cast<const float4*>(W)[i];
    ushort4 o;
    o.x = f2bf(w4.x); o.y = f2bf(w4.y); o.z = f2bf(w4.z); o.w = f2bf(w4.w);
    reinterpret_cast<ushort4*>(Wb)[i] = o;
}

// ---------------------------------------------------------------------------
// 3) spatial = sq @ W^T via v_wmma_f32_16x16x32_bf16, fused masked reduction.
//    Block = 256 threads = 8 waves; wave tile 32x64 (2x4 WMMA subtiles);
//    block tile 128(M) x 128(N); K loop over SDIM, 64 per outer step
//    (one 128B cacheline of bf16 per stream -> single prefetch per line).
//
//    Fragment layouts per CDNA5 ISA 7.12.2 (wave32):
//      A 16x32 bf16 : lane&15 = M row; lane>>4 selects K halves
//                     elems 0..7  -> K = 8*(lane>>4) + e
//                     elems 8..15 -> K = 16 + 8*(lane>>4) + (e-8)
//      B 32x16 bf16 : lane&15 = N col; lane>>4 selects K halves (same as A)
//                     elems 0..7 -> K = 8*(lane>>4) + e; 8..15 -> K = 16 + 8*(lane>>4) + (e-8)
//                     B[k][n] = W[n][k]  -> read row n of W, contiguous in k
//      C/D 16x16 f32: VGPR r, lane -> M = r + 8*(lane>>4), N = lane&15
// ---------------------------------------------------------------------------
__global__ __launch_bounds__(256)
void spatial_gemm_masked_sum(const unsigned short* __restrict__ sqB,
                             const unsigned short* __restrict__ Wb,
                             const float* __restrict__ y,
                             float* __restrict__ acc) {
    const int lane = threadIdx.x & 31;
    const int wave = threadIdx.x >> 5;
    const int half = lane >> 4;   // 0 or 1
    const int l15  = lane & 15;

    const int m0 = blockIdx.y * 128 + (wave >> 1) * 32;  // 4 waves along M
    const int n0 = blockIdx.x * 128 + (wave & 1) * 64;   // 2 waves along N

    v8f c[2][4] = {};

    const unsigned short* ap[2];
    ap[0] = sqB + (size_t)(m0 + l15)      * SDIM;
    ap[1] = sqB + (size_t)(m0 + 16 + l15) * SDIM;
    const unsigned short* bp[4];
#pragma unroll
    for (int j = 0; j < 4; ++j)
        bp[j] = Wb + (size_t)(n0 + 16 * j + l15) * SDIM + half * 8;

    for (int k = 0; k < SDIM; k += 64) {
        // one 128-byte line per stream, 256 bytes (two lines) ahead (global_prefetch_b8)
        __builtin_prefetch(ap[0] + k + 128, 0, 1);
        __builtin_prefetch(ap[1] + k + 128, 0, 1);
        __builtin_prefetch(bp[0] + k + 128, 0, 1);
        __builtin_prefetch(bp[1] + k + 128, 0, 1);
        __builtin_prefetch(bp[2] + k + 128, 0, 1);
        __builtin_prefetch(bp[3] + k + 128, 0, 1);

#pragma unroll
        for (int kk = k; kk < k + 64; kk += 32) {
            v16bf a[2], b[4];
#pragma unroll
            for (int i = 0; i < 2; ++i)
                a[i] = load_frag(ap[i] + kk + half * 8,
                                 ap[i] + kk + 16 + half * 8);
#pragma unroll
            for (int j = 0; j < 4; ++j)
                b[j] = load_frag(bp[j] + kk, bp[j] + kk + 16);

#pragma unroll
            for (int i = 0; i < 2; ++i)
#pragma unroll
                for (int j = 0; j < 4; ++j)
                    c[i][j] = __builtin_amdgcn_wmma_f32_16x16x32_bf16(
                        false, a[i], false, b[j], (short)0, c[i][j],
                        false, false);
        }
    }

    // masked reduction of the 32x64 wave tile; mask recomputed from y != 0
    float part = 0.f;
    const int rb = half * 8;
#pragma unroll
    for (int i = 0; i < 2; ++i) {
#pragma unroll
        for (int r = 0; r < 8; ++r) {
            const size_t row = (size_t)(m0 + 16 * i + rb + r) * SDIM;
#pragma unroll
            for (int j = 0; j < 4; ++j) {
                const int col = n0 + 16 * j + l15;
                part += c[i][j][r] * ((y[row + col] != 0.f) ? 1.f : 0.f);
            }
        }
    }
    for (int off = 16; off > 0; off >>= 1) part += __shfl_down(part, off, 32);

    __shared__ float sred[8];
    if (lane == 0) sred[wave] = part;
    __syncthreads();
    if (threadIdx.x == 0) {
        float s = 0.f;
        for (int w = 0; w < 8; ++w) s += sred[w];
        atomicAdd(acc + 0, s);
    }
}

// ---------------------------------------------------------------------------
// 4) loss = sqrt(total / mask_count + eps)
// ---------------------------------------------------------------------------
__global__ void finalize_loss(const float* __restrict__ acc,
                              float* __restrict__ out) {
    out[0] = sqrtf(acc[0] / acc[1] + EPSV);
}

// ---------------------------------------------------------------------------
// launcher
// ---------------------------------------------------------------------------
extern "C" void kernel_launch(void* const* d_in, const int* in_sizes, int n_in,
                              void* d_out, int out_size, void* d_ws, size_t ws_size,
                              hipStream_t stream) {
    const float* yhat = (const float*)d_in[0];
    const float* y    = (const float*)d_in[1];
    const float* W    = (const float*)d_in[2];
    float* out = (float*)d_out;

    char* ws = (char*)d_ws;
    unsigned short* sqB = (unsigned short*)ws;                             // 64 MB bf16 sq
    unsigned short* Wb  = (unsigned short*)(ws + (size_t)BDIM * SDIM * 2); // 32 MB bf16 W
    float* acc = (float*)(ws + (size_t)BDIM * SDIM * 2 + (size_t)SDIM * SDIM * 2);

    // acc[0] = masked total (sq + spatial), acc[1] = mask count
    hipMemsetAsync(acc, 0, 2 * sizeof(float), stream);

    {   // elementwise pass: one float4 per thread
        const int threads = 256;
        const int blocks = (int)(((size_t)BDIM * SDIM / 4) / threads);
        prep_sq_mask<<<blocks, threads, 0, stream>>>(yhat, y, sqB, acc);
    }
    {   // W -> bf16
        const int threads = 256;
        const int blocks = (int)(((size_t)SDIM * SDIM / 4) / threads);
        conv_w<<<blocks, threads, 0, stream>>>(W, Wb);
    }
    {   // WMMA GEMM + fused masked reduction
        dim3 grid(SDIM / 128, BDIM / 128);
        spatial_gemm_masked_sum<<<grid, 256, 0, stream>>>(sqB, Wb, y, acc);
    }
    finalize_loss<<<1, 1, 0, stream>>>(acc, out);
}
